// DisenTripletGCN_19000935317638
// MI455X (gfx1250) — matrix operand order repaired and checked by reading, not compiled, not run on hardware
//
#include <hip/hip_runtime.h>

typedef __attribute__((ext_vector_type(16))) _Float16 v16h;
typedef __attribute__((ext_vector_type(8)))  _Float16 h8;
typedef __attribute__((ext_vector_type(8)))  float    v8f;

#define LRELU(v) ((v) > 0.f ? (v) : 0.01f * (v))

// Get 32-bit LDS byte offset from a generic pointer to a __shared__ object.
__device__ __forceinline__ unsigned lds_off(const void* p) {
    return (unsigned)(unsigned long long)(__attribute__((address_space(3))) const void*)p;
}

// ---------------------------------------------------------------------------
// Tiled WMMA GEMM: C[M,N] = epilogue(A[M,K] (f16) x B[K,N] (f16) + bias)
// Block = 256 threads = 8 waves; block tile 128x128, K-step 32.
// B pre-transposed into K-panels: Bt[(k/32)*N + n][kk], kk = k%32.
// Tiles are staged with CDNA5 async global->LDS b128 loads into a 2-deep LDS
// double buffer: tile k+1 streams in (ASYNCcnt) while WMMAs consume tile k.
// Fragment layouts per CDNA5 ISA 7.12.2 (16-bit A 16x32, B 32x16, f32 C/D).
// ---------------------------------------------------------------------------
__global__ __launch_bounds__(256)
void k_gemm_wmma(const _Float16* __restrict__ A, const _Float16* __restrict__ Bt,
                 float* __restrict__ C,
                 const float* __restrict__ bias,
                 const float* __restrict__ bn_g, const float* __restrict__ bn_b,
                 int M, int N, int K)
{
    __shared__ __align__(16) _Float16 sA[2][128][40];  // [buf][row][kk] (+pad)
    __shared__ __align__(16) _Float16 sB[2][128][40];  // [buf][col][kk] (+pad)

    const int tid  = threadIdx.x;
    const int wave = tid >> 5;
    const int lane = tid & 31;
    const int m0 = blockIdx.y * 128;
    const int n0 = blockIdx.x * 128;

    v8f acc[8];
#pragma unroll
    for (int i = 0; i < 8; ++i)
#pragma unroll
        for (int j = 0; j < 8; ++j) acc[i][j] = 0.f;

    // staging coordinates: thread t covers 16 halves = 2x async b128
    const int srow = tid >> 1;          // 0..127 (A row / B col)
    const int scb  = (tid & 1) * 16;    // 0 or 16

    // Clamp OOB rows to M-1: they only feed outputs that are never stored,
    // and clamping keeps the per-wave async issue count uniform (4 per tile).
    const int grow = (m0 + srow < M) ? (m0 + srow) : (M - 1);
    const _Float16* gA = A + (size_t)grow * K + scb;

    auto issue = [&](int k0, int buf) {
        const unsigned la = lds_off(&sA[buf][srow][scb]);
        const unsigned lb = lds_off(&sB[buf][srow][scb]);
        const unsigned long long ga = (unsigned long long)(gA + k0);
        const unsigned long long gb = (unsigned long long)
            (Bt + ((size_t)(k0 >> 5) * N + n0 + srow) * 32 + scb);
        asm volatile("global_load_async_to_lds_b128 %0, %1, off\n\t"
                     "global_load_async_to_lds_b128 %0, %1, off offset:16"
                     :: "v"(la), "v"(ga) : "memory");
        asm volatile("global_load_async_to_lds_b128 %0, %1, off\n\t"
                     "global_load_async_to_lds_b128 %0, %1, off offset:16"
                     :: "v"(lb), "v"(gb) : "memory");
    };

    issue(0, 0);

    const int nk = K >> 5;
    for (int i = 0; i < nk; ++i) {
        const int buf = i & 1;
        if (i + 1 < nk) {
            issue((i + 1) << 5, buf ^ 1);
            // <=4 outstanding: (in-order) tile-i's 4 loads have completed.
            asm volatile("s_wait_asynccnt 0x4" ::: "memory");
        } else {
            asm volatile("s_wait_asynccnt 0x0" ::: "memory");
        }
        __syncthreads();   // all waves' tile-i LDS writes visible

        // A fragment: lane(0-15)=row M=lane, K = kb..kb+7 & kb+16..kb+23 (kb = 0|8)
        const int row = wave * 16 + (lane & 15);
        const int kb  = (lane >> 4) * 8;
        union { v16h v; h8 h[2]; } afrag;
        afrag.h[0] = *(const h8*)(&sA[buf][row][kb]);
        afrag.h[1] = *(const h8*)(&sA[buf][row][kb + 16]);

        // B fragment: lane(0-15)=col N=lane, K = kb2..kb2+7 & kb2+16..kb2+23 (kb2 = 0|8)
        const int kb2 = (lane >> 4) * 8;
        const int bc  = lane & 15;
#pragma unroll
        for (int nt = 0; nt < 8; ++nt) {
            union { v16h v; h8 h[2]; } bfrag;
            bfrag.h[0] = *(const h8*)(&sB[buf][nt * 16 + bc][kb2]);
            bfrag.h[1] = *(const h8*)(&sB[buf][nt * 16 + bc][kb2 + 16]);
            acc[nt] = __builtin_amdgcn_wmma_f32_16x16x32_f16(
                false, afrag.v, false, bfrag.v, (short)0, acc[nt], false, false);
        }
        __syncthreads();   // done with buf before it is refilled at i+2
    }

    // ---- epilogue: +bias [, *bn_g/sqrt(1+eps) + bn_b], leaky_relu(0.01) ----
    const float bnscale = rsqrtf(1.f + 1e-5f);
    const int crow_base = m0 + wave * 16 + (lane >> 4) * 8; // C: VGPR v -> M = base+v
    const int ccol_base = n0 + (lane & 15);
#pragma unroll
    for (int nt = 0; nt < 8; ++nt) {
        const int col = ccol_base + nt * 16;
        const float b = bias[col];
        float g = 0.f, bb = 0.f;
        if (bn_g) { g = bn_g[col] * bnscale; bb = bn_b[col]; }
#pragma unroll
        for (int v = 0; v < 8; ++v) {
            const int r = crow_base + v;
            if (r < M) {
                float val = acc[nt][v] + b;
                if (bn_g) val = val * g + bb;
                val = LRELU(val);
                C[(size_t)r * N + col] = val;
            }
        }
    }
}

// ---------------------------------------------------------------------------
// Per-capsule l2norm: wave per (row,capsule), 64 floats = float2 per lane.
// out = l2norm(in [+ addv]); optionally duplicate into copyv.
// ---------------------------------------------------------------------------
__global__ __launch_bounds__(256)
void k_caps_norm(const float* __restrict__ in, const float* __restrict__ addv,
                 float* __restrict__ out, float* __restrict__ copyv, long ntask)
{
    const long wid = ((long)blockIdx.x * blockDim.x + threadIdx.x) >> 5;
    const int lane = threadIdx.x & 31;
    if (wid >= ntask) return;
    float2 v = *((const float2*)(in + wid * 64) + lane);
    if (addv) {
        float2 a = *((const float2*)(addv + wid * 64) + lane);
        v.x += a.x; v.y += a.y;
    }
    float ss = v.x * v.x + v.y * v.y;
#pragma unroll
    for (int off = 16; off; off >>= 1) ss += __shfl_xor(ss, off, 32);
    const float inv = 1.f / fmaxf(sqrtf(ss), 1e-12f);
    float2 r; r.x = v.x * inv; r.y = v.y * inv;
    *((float2*)(out + wid * 64) + lane) = r;
    if (copyv) *((float2*)(copyv + wid * 64) + lane) = r;
}

// ---------------------------------------------------------------------------
// One routing iteration: wave per edge m.
//   z = x3[src[m]] (8x64), dot_k = <z_k, u[trg[m]]_k>, p = softmax_k(dot),
//   agg[trg[m]] += z * p    (fp32 global atomics)
// ---------------------------------------------------------------------------
__global__ __launch_bounds__(256)
void k_rout_iter(const float* __restrict__ x3, const float* __restrict__ u,
                 const int* __restrict__ edges, float* __restrict__ agg, int T)
{
    const long wid = ((long)blockIdx.x * blockDim.x + threadIdx.x) >> 5;
    const int lane = threadIdx.x & 31;
    if (wid >= T) return;
    const int s = edges[2 * wid];
    const int t = edges[2 * wid + 1];
    const float2* zp = (const float2*)(x3 + (size_t)s * 512);
    const float2* up = (const float2*)(u  + (size_t)t * 512);
    float2 z[8];
    float  dot[8];
#pragma unroll
    for (int k = 0; k < 8; ++k) {
        float2 zv = zp[k * 32 + lane];
        float2 uv = up[k * 32 + lane];
        z[k] = zv;
        float d = zv.x * uv.x + zv.y * uv.y;
#pragma unroll
        for (int off = 16; off; off >>= 1) d += __shfl_xor(d, off, 32);
        dot[k] = d;
    }
    float mx = dot[0];
#pragma unroll
    for (int k = 1; k < 8; ++k) mx = fmaxf(mx, dot[k]);
    float e[8], sum = 0.f;
#pragma unroll
    for (int k = 0; k < 8; ++k) { e[k] = __expf(dot[k] - mx); sum += e[k]; }
    const float invs = 1.f / sum;
    float* ap = agg + (size_t)t * 512;
#pragma unroll
    for (int k = 0; k < 8; ++k) {
        const float p = e[k] * invs;
        atomicAdd(ap + k * 64 + 2 * lane,     z[k].x * p);
        atomicAdd(ap + k * 64 + 2 * lane + 1, z[k].y * p);
    }
}

// Gather t_vecs = [obj[s] | pred | obj[o]] as f16. Block per edge, 128 threads.
__global__ __launch_bounds__(128)
void k_gather(const float* __restrict__ obj, const float* __restrict__ pred,
              const int* __restrict__ edges, _Float16* __restrict__ A1)
{
    const int m = blockIdx.x, j = threadIdx.x;
    const int s = edges[2 * m], o = edges[2 * m + 1];
    _Float16* r = A1 + (size_t)m * 384;
    r[j]       = (_Float16)obj[(size_t)s * 128 + j];
    r[128 + j] = (_Float16)pred[(size_t)m * 128 + j];
    r[256 + j] = (_Float16)obj[(size_t)o * 128 + j];
}

// Split out1 -> pred_out (d_out), atomically pool new_s/new_o + counts.
__global__ __launch_bounds__(256)
void k_split_pool(const float* __restrict__ out1, const int* __restrict__ edges,
                  float* __restrict__ pred_out, float* __restrict__ pooled,
                  float* __restrict__ counts)
{
    const int m = blockIdx.x, j = threadIdx.x;
    const int s = edges[2 * m], o = edges[2 * m + 1];
    const float* r = out1 + (size_t)m * 768;
    pred_out[(size_t)m * 256 + j] = r[256 + j];
    atomicAdd(&pooled[(size_t)s * 256 + j], r[j]);
    atomicAdd(&pooled[(size_t)o * 256 + j], r[512 + j]);
    if (j == 0) { atomicAdd(&counts[s], 1.f); atomicAdd(&counts[o], 1.f); }
}

__global__ void k_pool_div(const float* __restrict__ pooled,
                           const float* __restrict__ counts,
                           _Float16* __restrict__ A3, long n)
{
    const long i = (long)blockIdx.x * blockDim.x + threadIdx.x;
    if (i >= n) return;
    const float c = fmaxf(counts[i >> 8], 1.f);
    A3[i] = (_Float16)(pooled[i] / c);
}

__global__ void k_fill(float* __restrict__ p, float v, long n)
{
    const long stride = (long)gridDim.x * blockDim.x;
    for (long i = (long)blockIdx.x * blockDim.x + threadIdx.x; i < n; i += stride)
        p[i] = v;
}

__global__ void k_cast16(const float* __restrict__ s, _Float16* __restrict__ d, long n)
{
    const long stride = (long)gridDim.x * blockDim.x;
    for (long i = (long)blockIdx.x * blockDim.x + threadIdx.x; i < n; i += stride)
        d[i] = (_Float16)s[i];
}

// Cast weight B[K,N] (f32, row-major) into f16 K-panel layout:
// Bt[((k/32)*N + n)*32 + k%32]
__global__ void k_cast_wpanel(const float* __restrict__ B, _Float16* __restrict__ Bt,
                              int N, long total)
{
    const long i = (long)blockIdx.x * blockDim.x + threadIdx.x;
    if (i >= total) return;
    const int  kk = (int)(i & 31);
    const long r  = i >> 5;          // p*N + n
    const int  n  = (int)(r % N);
    const long p  = r / N;
    const long k  = p * 32 + kk;
    Bt[i] = (_Float16)B[k * N + n];
}

// ---------------------------------------------------------------------------
extern "C" void kernel_launch(void* const* d_in, const int* in_sizes, int n_in,
                              void* d_out, int out_size, void* d_ws, size_t ws_size,
                              hipStream_t stream)
{
    (void)n_in; (void)out_size; (void)ws_size; (void)in_sizes;
    const float* obj    = (const float*)d_in[0];
    const float* pred   = (const float*)d_in[1];
    const int*   edges  = (const int*)d_in[2];
    const float* pca_w1 = (const float*)d_in[3];
    const float* pca_b1 = (const float*)d_in[4];
    const float* clf_w1 = (const float*)d_in[5];
    const float* clf_b1 = (const float*)d_in[6];
    const float* bn_g1  = (const float*)d_in[7];
    const float* bn_b1  = (const float*)d_in[8];
    const float* pca_w2 = (const float*)d_in[9];
    const float* pca_b2 = (const float*)d_in[10];
    const float* clf_w2 = (const float*)d_in[11];
    const float* clf_b2 = (const float*)d_in[12];
    const float* bn_g2  = (const float*)d_in[13];
    const float* bn_b2  = (const float*)d_in[14];

    const int T = 80000, O = 20000;

    // ---- workspace carving ----
    char* base = (char*)d_ws;
    size_t off = 0;
    auto alloc = [&](size_t bytes) -> void* {
        void* p = base + off;
        off = (off + bytes + 255) & ~(size_t)255;
        return p;
    };
    _Float16* w1  = (_Float16*)alloc((size_t)384 * 512 * 2);
    _Float16* w2  = (_Float16*)alloc((size_t)512 * 768 * 2);
    _Float16* w3  = (_Float16*)alloc((size_t)256 * 512 * 2);
    _Float16* w4  = (_Float16*)alloc((size_t)512 * 256 * 2);
    _Float16* A16 = (_Float16*)alloc((size_t)T * 512 * 2);  // A1/A2/A3/A4 (sequential reuse)
    float* x      = (float*)alloc((size_t)T * 512 * 4);     // x after pca; later x2
    float* x3     = (float*)alloc((size_t)T * 512 * 4);
    float* agg    = (float*)alloc((size_t)T * 512 * 4);
    float* u      = (float*)alloc((size_t)T * 512 * 4);
    float* pooled = (float*)alloc((size_t)O * 256 * 4);
    float* counts = (float*)alloc((size_t)O * 4);
    float* out1   = x;  // alias over (dead) x..x3: needs T*768*4 <= 2*T*512*4

    float* obj_out  = (float*)d_out;                    // [O,256]
    float* pred_out = (float*)d_out + (size_t)O * 256;  // [T,256]

    const int FB = 2048; // grid-stride grid size

    // ---- cast weights to f16 K-panel layout ----
    k_cast_wpanel<<<(384 * 512 + 255) / 256, 256, 0, stream>>>(pca_w1, w1, 512, (long)384 * 512);
    k_cast_wpanel<<<(512 * 768 + 255) / 256, 256, 0, stream>>>(clf_w1, w2, 768, (long)512 * 768);
    k_cast_wpanel<<<(256 * 512 + 255) / 256, 256, 0, stream>>>(pca_w2, w3, 512, (long)256 * 512);
    k_cast_wpanel<<<(512 * 256 + 255) / 256, 256, 0, stream>>>(clf_w2, w4, 256, (long)512 * 256);

    // ---- stage 1: gather + pca GEMM ----
    k_gather<<<T, 128, 0, stream>>>(obj, pred, edges, A16);
    k_gemm_wmma<<<dim3(512 / 128, (T + 127) / 128), 256, 0, stream>>>(
        A16, w1, x, pca_b1, nullptr, nullptr, T, 512, 384);

    // ---- routing helper (runs on current buffers) ----
    auto routing = [&](int n) {
        const long caps = (long)n * 8;
        const int cgrid = (int)((caps * 32 + 255) / 256);
        const int rgrid = (n + 7) / 8;
        const float* xin = x;
        for (int layer = 0; layer < 2; ++layer) {
            k_caps_norm<<<cgrid, 256, 0, stream>>>(xin, nullptr, x3, u, caps);
            for (int it = 0; it < 3; ++it) {
                k_fill<<<FB, 256, 0, stream>>>(agg, 0.f, (long)n * 512);
                k_rout_iter<<<rgrid, 256, 0, stream>>>(x3, u, edges, agg, n);
                k_caps_norm<<<cgrid, 256, 0, stream>>>(agg, x3, u, nullptr, caps);
            }
            xin = u;
        }
    };
    routing(T);

    // ---- clf1 GEMM (+BN +lrelu) ----
    k_cast16<<<FB, 256, 0, stream>>>(u, A16, (long)T * 512);
    k_gemm_wmma<<<dim3(768 / 128, (T + 127) / 128), 256, 0, stream>>>(
        A16, w2, out1, clf_b1, bn_g1, bn_b1, T, 768, 512);

    // ---- split + mean pool ----
    k_fill<<<FB, 256, 0, stream>>>(pooled, 0.f, (long)O * 256);
    k_fill<<<(O + 255) / 256, 256, 0, stream>>>(counts, 0.f, O);
    k_split_pool<<<T, 256, 0, stream>>>(out1, edges, pred_out, pooled, counts);
    k_pool_div<<<(O * 256 + 255) / 256, 256, 0, stream>>>(pooled, counts, A16, (long)O * 256);

    // ---- stage 2 ----
    k_gemm_wmma<<<dim3(512 / 128, (O + 127) / 128), 256, 0, stream>>>(
        A16, w3, x, pca_b2, nullptr, nullptr, O, 512, 256);
    routing(O);
    k_cast16<<<FB, 256, 0, stream>>>(u, A16, (long)O * 512);
    k_gemm_wmma<<<dim3(256 / 128, (O + 127) / 128), 256, 0, stream>>>(
        A16, w4, obj_out, clf_b2, bn_g2, bn_b2, O, 256, 512);
}
